// MultiHeadAttention_63685775065692
// MI455X (gfx1250) — compile-verified, matrix operand order checked by reading, not run on hardware
//
#include <hip/hip_runtime.h>
#include <hip/hip_bf16.h>

typedef __attribute__((ext_vector_type(16))) _Float16 v16h;
typedef __attribute__((ext_vector_type(8)))  _Float16 v8h;
typedef __attribute__((ext_vector_type(8)))  float    v8f;
typedef __attribute__((ext_vector_type(4)))  unsigned v4u;
typedef __attribute__((ext_vector_type(8)))  int      v8i;
typedef __attribute__((ext_vector_type(4)))  int      v4i;

#define BW_TOT   1024           // B*W
#define SEQ_L    64
#define DIM      512
#define HEADS    8
#define HEAD_DIM 64
#define M_TOK    (BW_TOT * SEQ_L)   // 65536 token rows

#if defined(__HIP_DEVICE_COMPILE__) && __has_builtin(__builtin_amdgcn_tensor_load_to_lds)
#define HAVE_TDM 1
#endif

// ---------------------------------------------------------------------------
// Tensor Data Mover: 2D f16 tile (tile_rows x tile_cols) from a row-major
// tensor with row stride `row_stride` (elements) into contiguous LDS rows.
// D# layout per CDNA5 ISA ch.8 (group0: count/lds_addr/global_addr/type=2;
// group1: data_size=2B, tensor dims, tile dims, dim0 stride).
// ---------------------------------------------------------------------------
__device__ __forceinline__ void tdm_load_tile_f16(unsigned lds_byte_addr,
                                                  const _Float16* gptr,
                                                  unsigned tile_rows,
                                                  unsigned tile_cols,
                                                  unsigned row_stride,
                                                  unsigned tensor_rows)
{
#ifdef HAVE_TDM
    unsigned long long ga = (unsigned long long)(size_t)gptr;
    v4u g0;
    g0[0] = 1u;                                                // count=1 valid D#
    g0[1] = lds_byte_addr;                                     // lds_addr
    g0[2] = (unsigned)(ga & 0xFFFFFFFFu);                      // global_addr lo
    g0[3] = ((unsigned)((ga >> 32) & 0x01FFFFFFu)) | (2u << 30); // hi | type=2
    v8i g1;
    g1[0] = (int)(1u << 16);                                   // data_size=2B
    g1[1] = (int)((row_stride & 0xFFFFu) << 16);               // tensor_dim0 lo
    g1[2] = (int)(((row_stride >> 16) & 0xFFFFu) |
                  ((tensor_rows & 0xFFFFu) << 16));            // dim0 hi | dim1 lo
    g1[3] = (int)(((tensor_rows >> 16) & 0xFFFFu) |
                  ((tile_cols & 0xFFFFu) << 16));              // dim1 hi | tile_dim0
    g1[4] = (int)(tile_rows & 0xFFFFu);                        // tile_dim1 (dim2=0)
    g1[5] = (int)row_stride;                                   // tensor_dim0_stride lo
    g1[6] = 0;                                                 // stride hi | dim1_stride
    g1[7] = 0;
    v4i z4 = {0, 0, 0, 0};
#if __clang_major__ >= 23
    v8i z8 = {0, 0, 0, 0, 0, 0, 0, 0};
    __builtin_amdgcn_tensor_load_to_lds(g0, g1, z4, z4, z8, 0);
#else
    __builtin_amdgcn_tensor_load_to_lds(g0, g1, z4, z4, 0);
#endif
#else
    (void)lds_byte_addr; (void)gptr; (void)tile_rows;
    (void)tile_cols; (void)row_stride; (void)tensor_rows;
#endif
}

__device__ __forceinline__ void tdm_wait() {
#ifdef HAVE_TDM
    __builtin_amdgcn_s_wait_tensorcnt(0);
#endif
}

__device__ __forceinline__ unsigned lds_addr_of(const void* p) {
    return (unsigned)(size_t)p;   // generic LDS pointer: low 32 bits = LDS offset
}

// ---------------------------------------------------------------------------
// Fragment helpers (wave32 WMMA 16x16x32 f16 layouts, per CDNA5 ISA 7.12.2)
//   A (MxK): lane m = lane&15 ; halves -> K = bk..bk+7 , bk+16..bk+23 ,
//            bk = (lane>=16) ? 8 : 0
//   B (KxN): lane n = lane&15 ; same K mapping (store N-major in LDS)
//   C/D    : n = lane&15 ; m = vgpr + ((lane>=16) ? 8 : 0)
// ---------------------------------------------------------------------------
__device__ __forceinline__ v16h load_frag_rowmajor(const _Float16* base,
                                                   int row, int row_stride,
                                                   int koff, int bk) {
    const _Float16* p = base + row * row_stride + koff + bk;
    v8h lo = *(const v8h*)(p);
    v8h hi = *(const v8h*)(p + 16);
    return __builtin_shufflevector(lo, hi, 0,1,2,3,4,5,6,7,8,9,10,11,12,13,14,15);
}

// ---------------------------------------------------------------------------
// Tiled GEMM:  Out[M,N] = A[M,K] (f32 or f16) * Bw[K,N] (f32) + bias[N]
// 64x64 output tile per 256-thread block (8 waves, 2 WMMA tiles per wave).
// f16 A path stages its tile via the Tensor Data Mover.
// ---------------------------------------------------------------------------
template<bool A_IS_F16, bool OUT_F16>
__global__ __launch_bounds__(256)
void wmma_gemm_kernel(const void* __restrict__ Aptr,
                      const float* __restrict__ Bw,
                      const float* __restrict__ bias,
                      void* __restrict__ Out,
                      int M, int K, int N)
{
    __shared__ _Float16 ldsA[64 * 32];    // 64 rows x 32 K  (row-major)
    __shared__ _Float16 ldsBt[64 * 32];   // 64 cols x 32 K  (N-major)

    const int tid  = threadIdx.x;
    const int lane = tid & 31;
    const int wave = tid >> 5;
    const int half = lane >> 4;
    const int l15  = lane & 15;
    const int bk   = half * 8;

    const int m0 = blockIdx.x * 64;
    const int n0 = blockIdx.y * 64;

    const int mi      = (wave >> 1) * 16;   // this wave's tile row
    const int ni_base = (wave & 1) * 32;    // this wave's first tile col

    v8f acc[2] = {v8f{}, v8f{}};

    for (int k0 = 0; k0 < K; k0 += 32) {
        // ---- stage A tile (64x32) into LDS as f16 ----
        if (A_IS_F16) {
#ifdef HAVE_TDM
            if (tid == 0) {
                const _Float16* Ah = (const _Float16*)Aptr;
                tdm_load_tile_f16(lds_addr_of(ldsA),
                                  Ah + (size_t)m0 * K + k0,
                                  64, 32, (unsigned)K, (unsigned)M);
            }
#else
            const _Float16* Ah = (const _Float16*)Aptr;
            #pragma unroll
            for (int i = 0; i < 8; ++i) {
                int e = tid + i * 256;
                int r = e >> 5, c = e & 31;
                ldsA[r * 32 + c] = Ah[(size_t)(m0 + r) * K + k0 + c];
            }
#endif
        } else {
            const float* Af = (const float*)Aptr;
            #pragma unroll
            for (int i = 0; i < 8; ++i) {
                int e = tid + i * 256;            // 2048 elements
                int r = e >> 5, c = e & 31;
                ldsA[r * 32 + c] = (_Float16)Af[(size_t)(m0 + r) * K + k0 + c];
            }
        }
        // ---- stage B tile (32x64) into LDS transposed (N-major) ----
        #pragma unroll
        for (int i = 0; i < 8; ++i) {
            int e = tid + i * 256;
            int r = e >> 6, c = e & 63;           // r = K row, c = N col
            ldsBt[c * 32 + r] = (_Float16)Bw[(size_t)(k0 + r) * N + n0 + c];
        }
        // prefetch next K tile of B into GL2 (gfx1250 global_prefetch_b8)
        if (k0 + 32 < K) {
            __builtin_prefetch(&Bw[(size_t)(k0 + 32 + (tid >> 6)) * N + n0 + (tid & 63)], 0, 1);
        }
        if (A_IS_F16) tdm_wait();                 // TENSORcnt -> 0 before barrier
        __syncthreads();

        v16h afrag = load_frag_rowmajor(ldsA, mi + l15, 32, 0, bk);
        #pragma unroll
        for (int r = 0; r < 2; ++r) {
            int ni = ni_base + r * 16;
            v16h bfrag = load_frag_rowmajor(ldsBt, ni + l15, 32, 0, bk);
            acc[r] = __builtin_amdgcn_wmma_f32_16x16x32_f16(
                false, afrag, false, bfrag, (short)0, acc[r], false, false);
        }
        __syncthreads();
    }

    // ---- writeback with bias ----
    #pragma unroll
    for (int r = 0; r < 2; ++r) {
        int ni = ni_base + r * 16;
        int n  = n0 + ni + l15;
        float bv = bias ? bias[n] : 0.0f;
        #pragma unroll
        for (int t = 0; t < 8; ++t) {
            int m = m0 + mi + half * 8 + t;
            float val = acc[r][t] + bv;
            if (OUT_F16)
                ((_Float16*)Out)[(size_t)m * N + n] = (_Float16)val;
            else
                ((float*)Out)[(size_t)m * N + n] = val;
        }
    }
}

// ---------------------------------------------------------------------------
// Attention: one block per (bw, head). q,k,v head tiles are 64x64 f16.
// q/k tiles staged by the Tensor Data Mover; v staged manually (transposed).
// logits = q k^T * 1/8 + alibi(m,n) ; mask ; softmax ; o = p v
// ---------------------------------------------------------------------------
__global__ __launch_bounds__(256)
void wmma_attention_kernel(const _Float16* __restrict__ qg,
                           const _Float16* __restrict__ kg,
                           const _Float16* __restrict__ vg,
                           const unsigned char* __restrict__ mask,
                           _Float16* __restrict__ og)
{
    __shared__ _Float16 sQ[64 * 64];     // query-major
    __shared__ _Float16 sK[64 * 64];     // key-major   (B fragment for QK^T)
    __shared__ _Float16 sVt[64 * 64];    // hd-major    (B fragment for PV)
    __shared__ float    sS[64 * 64];     // logits f32
    __shared__ _Float16 sP[64 * 64];     // softmax probs f16

    const int bw = blockIdx.x;
    const int h  = blockIdx.y;

    const int tid  = threadIdx.x;
    const int lane = tid & 31;
    const int wave = tid >> 5;
    const int half = lane >> 4;
    const int l15  = lane & 15;
    const int bk   = half * 8;

    const int mi      = (wave >> 1) * 16;
    const int ni_base = (wave & 1) * 32;

    const size_t base = (size_t)bw * SEQ_L * DIM + (size_t)h * HEAD_DIM;

#ifdef HAVE_TDM
    // ---- TDM: async-stage q and k head tiles (64 rows x 64 f16, stride 512) ----
    if (tid == 0) {
        tdm_load_tile_f16(lds_addr_of(sQ), qg + base, SEQ_L, HEAD_DIM, DIM, M_TOK);
        tdm_load_tile_f16(lds_addr_of(sK), kg + base, SEQ_L, HEAD_DIM, DIM, M_TOK);
    }
#else
    #pragma unroll
    for (int i = 0; i < 2; ++i) {
        int c   = tid + i * 256;          // 512 chunks of 8 halves
        int row = c >> 3, c8 = c & 7;
        *(float4*)&sQ[row * 64 + c8 * 8] =
            *(const float4*)&qg[base + (size_t)row * DIM + c8 * 8];
        *(float4*)&sK[row * 64 + c8 * 8] =
            *(const float4*)&kg[base + (size_t)row * DIM + c8 * 8];
    }
#endif
    // ---- v staged transposed (hd-major) while TDM copies are in flight ----
    #pragma unroll
    for (int i = 0; i < 16; ++i) {
        int e   = tid + i * 256;          // 4096 scalars
        int row = e >> 6, col = e & 63;   // row = key, col = hd
        sVt[col * 64 + row] = vg[base + (size_t)row * DIM + col];
    }
    if (tid == 0) tdm_wait();             // TENSORcnt -> 0 in issuing wave
    __syncthreads();

    // ---- logits = q k^T ----
    v8f acc[2] = {v8f{}, v8f{}};
    #pragma unroll
    for (int k0 = 0; k0 < HEAD_DIM; k0 += 32) {
        v16h afrag = load_frag_rowmajor(sQ, mi + l15, 64, k0, bk);
        #pragma unroll
        for (int r = 0; r < 2; ++r) {
            int ni = ni_base + r * 16;
            v16h bfrag = load_frag_rowmajor(sK, ni + l15, 64, k0, bk);
            acc[r] = __builtin_amdgcn_wmma_f32_16x16x32_f16(
                false, afrag, false, bfrag, (short)0, acc[r], false, false);
        }
    }

    // ---- scale + alibi + mask -> sS ----
    const float scale = 0.125f;                     // 1/sqrt(64)
    const float slope = powf(10.0f, -(float)h / 7.0f);
    const unsigned char* mrow = mask + (size_t)bw * SEQ_L * SEQ_L;
    #pragma unroll
    for (int r = 0; r < 2; ++r) {
        int n  = ni_base + r * 16 + l15;
        int kx = n >> 3, ky = n & 7;                // window coords (ws=8)
        #pragma unroll
        for (int t = 0; t < 8; ++t) {
            int m  = mi + half * 8 + t;
            int qx = m >> 3, qy = m & 7;
            int dist = (qx > kx ? qx - kx : kx - qx) + (qy > ky ? qy - ky : ky - qy);
            float val = acc[r][t] * scale - slope * (float)dist;
            if (!mrow[m * SEQ_L + n]) val = -1e30f;
            sS[m * 64 + n] = val;
        }
    }
    __syncthreads();

    // ---- row softmax (64 rows, one thread each) ----
    if (tid < 64) {
        float mx = -1e30f;
        for (int j = 0; j < 64; ++j) mx = fmaxf(mx, sS[tid * 64 + j]);
        float sum = 0.0f;
        for (int j = 0; j < 64; ++j) sum += __expf(sS[tid * 64 + j] - mx);
        float inv = 1.0f / sum;
        for (int j = 0; j < 64; ++j)
            sP[tid * 64 + j] = (_Float16)(__expf(sS[tid * 64 + j] - mx) * inv);
    }
    __syncthreads();

    // ---- o = p v ----
    v8f occ[2] = {v8f{}, v8f{}};
    #pragma unroll
    for (int k0 = 0; k0 < SEQ_L; k0 += 32) {
        v16h afrag = load_frag_rowmajor(sP, mi + l15, 64, k0, bk);
        #pragma unroll
        for (int r = 0; r < 2; ++r) {
            int ni = ni_base + r * 16;
            v16h bfrag = load_frag_rowmajor(sVt, ni + l15, 64, k0, bk);
            occ[r] = __builtin_amdgcn_wmma_f32_16x16x32_f16(
                false, afrag, false, bfrag, (short)0, occ[r], false, false);
        }
    }
    #pragma unroll
    for (int r = 0; r < 2; ++r) {
        int n = ni_base + r * 16 + l15;               // hd column
        #pragma unroll
        for (int t = 0; t < 8; ++t) {
            int m = mi + half * 8 + t;                // query row
            og[base + (size_t)m * DIM + n] = (_Float16)occ[r][t];
        }
    }
}

// ---------------------------------------------------------------------------
// Host launcher
// ---------------------------------------------------------------------------
extern "C" void kernel_launch(void* const* d_in, const int* in_sizes, int n_in,
                              void* d_out, int out_size, void* d_ws, size_t ws_size,
                              hipStream_t stream) {
    (void)in_sizes; (void)n_in; (void)out_size; (void)ws_size;

    const float* x_q  = (const float*)d_in[0];
    const float* x_kv = (const float*)d_in[1];
    const unsigned char* mask = (const unsigned char*)d_in[2];
    const float* Wq = (const float*)d_in[3];
    const float* bq = (const float*)d_in[4];
    const float* Wk = (const float*)d_in[5];
    const float* bk = (const float*)d_in[6];
    const float* Wv = (const float*)d_in[7];
    const float* bv = (const float*)d_in[8];
    const float* Wo = (const float*)d_in[9];
    const float* bo = (const float*)d_in[10];

    // workspace: q,k,v,o in f16  (4 x 64 MB)
    _Float16* qf = (_Float16*)d_ws;
    _Float16* kf = qf + (size_t)M_TOK * DIM;
    _Float16* vf = kf + (size_t)M_TOK * DIM;
    _Float16* of = vf + (size_t)M_TOK * DIM;

    dim3 gemm_grid(M_TOK / 64, DIM / 64);   // 1024 x 8
    dim3 block(256);

    // Q/K/V projections (f32 in -> f16 out)
    wmma_gemm_kernel<false, true><<<gemm_grid, block, 0, stream>>>(
        (const void*)x_q, Wq, bq, (void*)qf, M_TOK, DIM, DIM);
    wmma_gemm_kernel<false, true><<<gemm_grid, block, 0, stream>>>(
        (const void*)x_kv, Wk, bk, (void*)kf, M_TOK, DIM, DIM);
    wmma_gemm_kernel<false, true><<<gemm_grid, block, 0, stream>>>(
        (const void*)x_kv, Wv, bv, (void*)vf, M_TOK, DIM, DIM);

    // attention per (bw, head)
    dim3 attn_grid(BW_TOT, HEADS);          // 1024 x 8
    wmma_attention_kernel<<<attn_grid, block, 0, stream>>>(qf, kf, vf, mask, of);

    // output projection (f16 in -> f32 out)
    wmma_gemm_kernel<true, false><<<gemm_grid, block, 0, stream>>>(
        (const void*)of, Wo, bo, d_out, M_TOK, DIM, DIM);
}
